// EdgeUpdate_11038065950784
// MI455X (gfx1250) — matrix operand order repaired and checked by reading, not compiled, not run on hardware
//
#include <hip/hip_runtime.h>
#include <hip/hip_bf16.h>

// ---------------------------------------------------------------------------
// CDNA5 (gfx1250) fused edge-MLP:
//   h = silu(concat(src,dst,edge,rbf) @ W1 + b1); h = silu(h @ W2 + b2)
//   out = LN(edge + h)
// Both GEMMs run on v_wmma_f32_16x16x32_bf16 (wave32).
// ---------------------------------------------------------------------------

typedef __attribute__((ext_vector_type(16))) __bf16 v16bf;
typedef __attribute__((ext_vector_type(8)))  __bf16 v8bf;
typedef __attribute__((ext_vector_type(8)))  float  v8f;

#define C_NODE 128
#define C_EDGE 128
#define RBF    16
#define K1     416            // 400 padded to 13*32
#define K1PAD  424            // LDS row stride for A (keeps 16B alignment: 424*2=848=16*53)
#define K2PAD  136            // LDS row stride for H (136*2=272=16*17)
#define TILE_M 32             // edges per block
#define NW1T   (128 * K1)     // bf16 elements of transposed padded W1 in ws

union V16U { v16bf v; v8bf h[2]; };

__device__ __forceinline__ float silu_f(float x) {
    return x * (1.0f / (1.0f + __expf(-x)));
}

// ---------------------------------------------------------------------------
// Kernel 0: prepack weights into bf16, transposed, K-padded, into d_ws.
//   ws[0 .. NW1T)             : W1t[n][k]  (n=0..127, k=0..415; k>=400 -> 0)
//   ws[NW1T .. NW1T+128*128)  : W2t[n][k]
// ---------------------------------------------------------------------------
__global__ __launch_bounds__(256)
void edgemlp_prepack_kernel(const float* __restrict__ W1,
                            const float* __restrict__ W2,
                            __bf16* __restrict__ ws) {
    int i = blockIdx.x * 256 + threadIdx.x;
    if (i < NW1T) {
        int n = i / K1, k = i % K1;
        float v = (k < 400) ? W1[k * C_EDGE + n] : 0.0f;
        ws[i] = (__bf16)v;
    } else if (i < NW1T + 128 * 128) {
        int j = i - NW1T;
        int n = j >> 7, k = j & 127;
        ws[NW1T + j] = (__bf16)W2[k * C_EDGE + n];
    }
}

// ---------------------------------------------------------------------------
// Kernel 1: fused edge update. 256 threads = 8 waves; TILE_M=32 edges/block.
// Wave w owns output columns [16w, 16w+16); loops over 2 M-tiles of 16 rows.
// ---------------------------------------------------------------------------
__global__ __launch_bounds__(256)
void edgemlp_fused_kernel(const float* __restrict__ node_scalars,
                          const float* __restrict__ edge_feats,
                          const float* __restrict__ rbf,
                          const int*   __restrict__ src_idxs,
                          const int*   __restrict__ dst_idxs,
                          const __bf16* __restrict__ W1t,   // [128][416] bf16
                          const __bf16* __restrict__ W2t,   // [128][128] bf16
                          const float* __restrict__ b1,
                          const float* __restrict__ b2,
                          const float* __restrict__ gamma,
                          const float* __restrict__ beta,
                          float* __restrict__ out,
                          int n_edges) {
    __shared__ __align__(32) __bf16 sA[TILE_M * K1PAD];   // 27,136 B
    __shared__ __align__(32) __bf16 sH[TILE_M * K2PAD];   //  8,704 B
    __shared__ __align__(32) float  sY[TILE_M * 132];     // 16,896 B

    const int tid  = threadIdx.x;
    const int wave = tid >> 5;          // 0..7  -> output column tile
    const int lane = tid & 31;
    const int c    = lane & 15;         // column within 16-wide tile
    const int hk   = lane >> 4;         // K-half select (WMMA lane layout)
    const int n0   = wave * 16;
    const int e0   = blockIdx.x * TILE_M;

    // ---- Stage 0: build bf16 A tile [32 x 416] = concat(src,dst,edge,rbf,0)
    for (int idx = tid; idx < TILE_M * K1; idx += 256) {
        int r = idx / K1, k = idx % K1;
        int e = e0 + r;
        if (e >= n_edges) e = 0;        // clamp (no tail with E%32==0, but safe)
        float v;
        if (k < C_NODE) {
            v = node_scalars[(size_t)src_idxs[e] * C_NODE + k];
        } else if (k < 2 * C_NODE) {
            v = node_scalars[(size_t)dst_idxs[e] * C_NODE + (k - C_NODE)];
        } else if (k < 2 * C_NODE + C_EDGE) {
            v = edge_feats[(size_t)e * C_EDGE + (k - 2 * C_NODE)];
        } else if (k < 2 * C_NODE + C_EDGE + RBF) {
            v = rbf[(size_t)e * RBF + (k - 2 * C_NODE - C_EDGE)];
        } else {
            v = 0.0f;                   // K padding 400..415
        }
        sA[r * K1PAD + k] = (__bf16)v;
    }
    __syncthreads();

    // ---- Stage 1: GEMM1 (K=416, 13 chunks of 32) -> silu -> bf16 H in LDS
    v8f acc0 = {}, acc1 = {};
    const __bf16* w1col = W1t + (size_t)(n0 + c) * K1;
    for (int kc = 0; kc < 13; ++kc) {
        const int kb = kc * 32;
        // B fragment: lane = column (n0+c); halves at kb+8*hk and kb+16+8*hk
        V16U bu;
        bu.h[0] = *(const v8bf*)(w1col + kb + 8 * hk);
        bu.h[1] = *(const v8bf*)(w1col + kb + 16 + 8 * hk);
        v16bf bfrag = bu.v;
        // A fragments: lane = row (m*16 + c); halves at kb+8*hk and kb+16+8*hk
        V16U a0, a1;
        const __bf16* ar0 = sA + (0 * 16 + c) * K1PAD;
        const __bf16* ar1 = sA + (1 * 16 + c) * K1PAD;
        a0.h[0] = *(const v8bf*)(ar0 + kb + 8 * hk);
        a0.h[1] = *(const v8bf*)(ar0 + kb + 16 + 8 * hk);
        a1.h[0] = *(const v8bf*)(ar1 + kb + 8 * hk);
        a1.h[1] = *(const v8bf*)(ar1 + kb + 16 + 8 * hk);
        acc0 = __builtin_amdgcn_wmma_f32_16x16x32_bf16(
                   false, a0.v, false, bfrag, (short)0, acc0, false, false);
        acc1 = __builtin_amdgcn_wmma_f32_16x16x32_bf16(
                   false, a1.v, false, bfrag, (short)0, acc1, false, false);
    }
    {
        const float bias1 = b1[n0 + c];
        #pragma unroll
        for (int i = 0; i < 8; ++i) {
            int rr = i + 8 * hk;        // row within 16-tile (C/D lane layout)
            float x0 = silu_f(acc0[i] + bias1);
            float x1 = silu_f(acc1[i] + bias1);
            sH[(rr)      * K2PAD + n0 + c] = (__bf16)x0;
            sH[(rr + 16) * K2PAD + n0 + c] = (__bf16)x1;
        }
    }
    __syncthreads();

    // ---- Stage 2: GEMM2 (K=128, 4 chunks) -> silu -> +residual -> fp32 Y
    v8f d0 = {}, d1 = {};
    const __bf16* w2col = W2t + (size_t)(n0 + c) * C_EDGE;
    for (int kc = 0; kc < 4; ++kc) {
        const int kb = kc * 32;
        V16U bu;
        bu.h[0] = *(const v8bf*)(w2col + kb + 8 * hk);
        bu.h[1] = *(const v8bf*)(w2col + kb + 16 + 8 * hk);
        v16bf bfrag = bu.v;
        V16U a0, a1;
        const __bf16* hr0 = sH + (0 * 16 + c) * K2PAD;
        const __bf16* hr1 = sH + (1 * 16 + c) * K2PAD;
        a0.h[0] = *(const v8bf*)(hr0 + kb + 8 * hk);
        a0.h[1] = *(const v8bf*)(hr0 + kb + 16 + 8 * hk);
        a1.h[0] = *(const v8bf*)(hr1 + kb + 8 * hk);
        a1.h[1] = *(const v8bf*)(hr1 + kb + 16 + 8 * hk);
        d0 = __builtin_amdgcn_wmma_f32_16x16x32_bf16(
                 false, a0.v, false, bfrag, (short)0, d0, false, false);
        d1 = __builtin_amdgcn_wmma_f32_16x16x32_bf16(
                 false, a1.v, false, bfrag, (short)0, d1, false, false);
    }
    {
        const float bias2 = b2[n0 + c];
        #pragma unroll
        for (int i = 0; i < 8; ++i) {
            int rr = i + 8 * hk;
            int e_lo = e0 + rr, e_hi = e0 + rr + 16;
            float x0 = silu_f(d0[i] + bias2);
            float x1 = silu_f(d1[i] + bias2);
            float r0 = (e_lo < n_edges) ? edge_feats[(size_t)e_lo * C_EDGE + n0 + c] : 0.0f;
            float r1 = (e_hi < n_edges) ? edge_feats[(size_t)e_hi * C_EDGE + n0 + c] : 0.0f;
            sY[(rr)      * 132 + n0 + c] = r0 + x0;
            sY[(rr + 16) * 132 + n0 + c] = r1 + x1;
        }
    }
    __syncthreads();

    // ---- Stage 3: LayerNorm over 128 cols; 8 lanes per row, wave32 shuffle
    {
        const int r    = tid >> 3;          // 0..31 (4 rows per wave)
        const int part = tid & 7;           // 16 cols per lane
        const float* yr = sY + r * 132 + part * 16;
        float vals[16];
        float s = 0.0f, ss = 0.0f;
        #pragma unroll
        for (int j = 0; j < 16; ++j) {
            float v = yr[j];
            vals[j] = v;
            s  += v;
            ss += v * v;
        }
        #pragma unroll
        for (int off = 1; off < 8; off <<= 1) {
            s  += __shfl_xor(s,  off, 32);
            ss += __shfl_xor(ss, off, 32);
        }
        const float mu   = s * (1.0f / 128.0f);
        const float var  = ss * (1.0f / 128.0f) - mu * mu;
        const float rstd = rsqrtf(var + 1e-5f);
        const int e = e0 + r;
        if (e < n_edges) {
            float* orow = out + (size_t)e * C_EDGE + part * 16;
            #pragma unroll
            for (int j = 0; j < 16; ++j) {
                int col = part * 16 + j;
                orow[j] = (vals[j] - mu) * rstd * gamma[col] + beta[col];
            }
        }
    }
}

// ---------------------------------------------------------------------------
extern "C" void kernel_launch(void* const* d_in, const int* in_sizes, int n_in,
                              void* d_out, int out_size, void* d_ws, size_t ws_size,
                              hipStream_t stream) {
    const float* node_scalars = (const float*)d_in[0];
    const float* edge_feats   = (const float*)d_in[1];
    const float* rbf          = (const float*)d_in[2];
    const int*   src_idxs     = (const int*)d_in[3];
    const int*   dst_idxs     = (const int*)d_in[4];
    const float* W1           = (const float*)d_in[5];
    const float* b1           = (const float*)d_in[6];
    const float* W2           = (const float*)d_in[7];
    const float* b2           = (const float*)d_in[8];
    const float* gamma        = (const float*)d_in[9];
    const float* beta         = (const float*)d_in[10];
    float* out = (float*)d_out;

    const int n_edges = in_sizes[3];

    __bf16* ws_bf = (__bf16*)d_ws;
    const __bf16* W1t = ws_bf;
    const __bf16* W2t = ws_bf + NW1T;

    // Kernel 0: weight prepack (bf16, transposed, K-padded)
    const int prep_elems = NW1T + 128 * 128;
    edgemlp_prepack_kernel<<<(prep_elems + 255) / 256, 256, 0, stream>>>(W1, W2, ws_bf);

    // Kernel 1: fused edge MLP
    const int nblocks = (n_edges + TILE_M - 1) / TILE_M;
    edgemlp_fused_kernel<<<nblocks, 256, 0, stream>>>(
        node_scalars, edge_feats, rbf, src_idxs, dst_idxs,
        W1t, W2t, b1, b2, gamma, beta, out, n_edges);
}
